// AttnEncoder_63307817943740
// MI455X (gfx1250) — matrix operand order repaired and checked by reading, not compiled, not run on hardware
//
#include <hip/hip_runtime.h>
#include <hip/hip_bf16.h>

// ---------------------------------------------------------------------------
// Attention-gated LSTM for MI455X (gfx1250, wave32, WMMA).
// B=4096, T=64, D=H=128. Each workgroup owns 16 batch rows for all 64 steps.
// Key change vs v1: all 32 weight B-fragments (16x16x32 bf16) live in VGPRs
// for the whole time loop (256 VGPRs/wave; gfx1250 allows 1024), so the
// steady state is only A-fragment LDS reads + 32 WMMAs per wave per step.
// ---------------------------------------------------------------------------

#define B_SZ   4096
#define T_SZ   64
#define D_SZ   128
#define H_SZ   128
#define NG     512      // 4*H gate columns
#define KC     256      // concat K = D + H
#define ROWS   16       // batch rows per workgroup
#define NTHR   256      // 8 wave32s
#define KPAD   264      // padded bf16 row stride (528 bytes) -> conflict-free

typedef __bf16 bf16_t;
typedef __attribute__((ext_vector_type(16))) __bf16 v16bf;
typedef __attribute__((ext_vector_type(8)))  __bf16 v8bf;
typedef __attribute__((ext_vector_type(8)))  float  v8f;

#define SMEM_BYTES (ROWS * KPAD * 2 + 3 * ROWS * H_SZ * 4)   // ~33 KB

__device__ __forceinline__ unsigned short f32_to_bf16_rne(float f) {
    unsigned int u = __builtin_bit_cast(unsigned int, f);
    u += 0x7FFFu + ((u >> 16) & 1u);
    return (unsigned short)(u >> 16);
}

__device__ __forceinline__ bf16_t f2bf(float f) {
    return __builtin_bit_cast(bf16_t, f32_to_bf16_rne(f));
}

__device__ __forceinline__ float fast_sigmoid(float x) {
    return 1.0f / (1.0f + __expf(-x));
}

__device__ __forceinline__ float fast_tanh(float x) {
    float e = __expf(2.0f * x);
    return 1.0f - 2.0f / (e + 1.0f);
}

__global__ __launch_bounds__(NTHR, 1)
void attn_lstm_fused(const float* __restrict__ x,       // [B][T][D]
                     const float* __restrict__ W_ih,    // [NG][D]
                     const float* __restrict__ W_hh,    // [NG][H]
                     const float* __restrict__ b_ih,    // [NG]
                     const float* __restrict__ b_hh,    // [NG]
                     const float* __restrict__ W_attn,  // [2H + T]
                     const float* __restrict__ b_attn,  // [1]
                     float* __restrict__ out)           // [B][T][H]
{
    extern __shared__ char smem[];
    unsigned short* sA  = (unsigned short*)smem;              // [ROWS][KPAD] bf16 [w_in | h]
    float*          sH  = (float*)(sA + ROWS * KPAD);         // [ROWS][H]
    float*          sC  = sH + ROWS * H_SZ;                   // [ROWS][H]
    float*          sXT = sC + ROWS * H_SZ;                   // [ROWS][D] x_time tile

    const int tid   = threadIdx.x;
    const int wave  = tid >> 5;
    const int lane  = tid & 31;
    const int lan15 = lane & 15;
    const int hi    = lane >> 4;          // half-wave select per WMMA layout
    const int b0    = blockIdx.x * ROWS;

    // =====================================================================
    // Preload ALL weight B-fragments into registers (loop-invariant).
    // B frag for (k-step kk, tile p): lane holds column n = (wave+8p)*16+lan15,
    // 16 consecutive k at kbase = kk*32 + 16*hi.  gates[m,n] = sum_k A[m,k]*W[n,k]
    // with k<128 -> W_ih[n][k], k>=128 -> W_hh[n][k-128] (chunks never straddle).
    // =====================================================================
    v16bf Bfr[8][4];
    #pragma unroll
    for (int kk = 0; kk < 8; ++kk) {
        const int kbase = kk * 32 + hi * 16;
        const float* src = (kbase < D_SZ) ? (W_ih + kbase) : (W_hh + (kbase - D_SZ));
        #pragma unroll
        for (int p = 0; p < 4; ++p) {
            const int n = (wave + 8 * p) * 16 + lan15;
            const float* sp = src + n * D_SZ;    // 16 contiguous f32 (64 B)
            v16bf f;
            #pragma unroll
            for (int e = 0; e < 16; ++e) f[e] = f2bf(sp[e]);
            Bfr[kk][p] = f;
        }
    }

    // ---- init h, c and compute x_time tile (fused einsum 'btd,t->bd')
    for (int idx = tid; idx < ROWS * D_SZ; idx += NTHR) {
        int r = idx >> 7;
        int j = idx & (D_SZ - 1);
        const float* xp = x + ((long)(b0 + r) * T_SZ) * D_SZ + j;
        float acc = 0.0f;
        #pragma unroll 4
        for (int t = 0; t < T_SZ; ++t)
            acc += xp[t * D_SZ] * W_attn[2 * H_SZ + t];   // wt[t]: uniform scalar loads
        sXT[idx] = acc;
        sH[idx] = 0.0f;
        sC[idx] = 0.0f;
    }

    // ---- per-lane persistent attention weights (4 elems at j = 4*lane)
    const int j4 = lane * 4;
    const float wh0 = W_attn[j4 + 0], wh1 = W_attn[j4 + 1];
    const float wh2 = W_attn[j4 + 2], wh3 = W_attn[j4 + 3];
    const float wc0 = W_attn[H_SZ + j4 + 0], wc1 = W_attn[H_SZ + j4 + 1];
    const float wc2 = W_attn[H_SZ + j4 + 2], wc3 = W_attn[H_SZ + j4 + 3];
    const float battn = b_attn[0];

    // ---- per-lane gate biases: wave w + lane column covers i/f/g/o of h-col colw
    const int colw = wave * 16 + lan15;
    const float bi  = b_ih[colw]       + b_hh[colw];
    const float bff = b_ih[colw + 128] + b_hh[colw + 128];
    const float bg  = b_ih[colw + 256] + b_hh[colw + 256];
    const float bo  = b_ih[colw + 384] + b_hh[colw + 384];

    __syncthreads();

    for (int t = 0; t < T_SZ; ++t) {
        // ============ phase 1: attention softmax + A = [w_in | h] (bf16) ==========
        #pragma unroll
        for (int rr = 0; rr < 2; ++rr) {
            const int row = wave * 2 + rr;
            const float* hrow = sH + row * H_SZ;
            const float* crow = sC + row * H_SZ;
            float h0 = hrow[j4], h1 = hrow[j4 + 1], h2 = hrow[j4 + 2], h3 = hrow[j4 + 3];
            float c0 = crow[j4], c1 = crow[j4 + 1], c2 = crow[j4 + 2], c3 = crow[j4 + 3];

            float part = h0 * wh0 + h1 * wh1 + h2 * wh2 + h3 * wh3
                       + c0 * wc0 + c1 * wc1 + c2 * wc2 + c3 * wc3;
            #pragma unroll
            for (int m = 16; m > 0; m >>= 1) part += __shfl_xor(part, m, 32);
            const float sbase = part + battn;

            const float* xtrow = sXT + row * D_SZ;
            float s0 = xtrow[j4] + sbase,     s1 = xtrow[j4 + 1] + sbase;
            float s2 = xtrow[j4 + 2] + sbase, s3 = xtrow[j4 + 3] + sbase;

            float mx = fmaxf(fmaxf(s0, s1), fmaxf(s2, s3));
            #pragma unroll
            for (int m = 16; m > 0; m >>= 1) mx = fmaxf(mx, __shfl_xor(mx, m, 32));

            float e0 = __expf(s0 - mx), e1 = __expf(s1 - mx);
            float e2 = __expf(s2 - mx), e3 = __expf(s3 - mx);
            float sum = e0 + e1 + e2 + e3;
            #pragma unroll
            for (int m = 16; m > 0; m >>= 1) sum += __shfl_xor(sum, m, 32);
            const float inv = 1.0f / sum;

            const float* xp = x + ((long)(b0 + row) * T_SZ + t) * D_SZ + j4;
            float xt0 = xp[0], xt1 = xp[1], xt2 = xp[2], xt3 = xp[3];
            __builtin_prefetch(xp + D_SZ, 0, 3);   // next timestep slice -> global_prefetch

            unsigned short* arow = sA + row * KPAD;
            ushort4 wq;
            wq.x = f32_to_bf16_rne(e0 * inv * xt0);
            wq.y = f32_to_bf16_rne(e1 * inv * xt1);
            wq.z = f32_to_bf16_rne(e2 * inv * xt2);
            wq.w = f32_to_bf16_rne(e3 * inv * xt3);
            *(ushort4*)(arow + j4) = wq;                     // w_in -> k [0,128)

            ushort4 hq;
            hq.x = f32_to_bf16_rne(h0); hq.y = f32_to_bf16_rne(h1);
            hq.z = f32_to_bf16_rne(h2); hq.w = f32_to_bf16_rne(h3);
            *(ushort4*)(arow + D_SZ + j4) = hq;              // h -> k [128,256)
        }
        __syncthreads();

        // ============ phase 2: gates = A * Wcat, WMMA with register-resident B =====
        v8f acc0 = {0,0,0,0,0,0,0,0};   // i : cols 16w
        v8f acc1 = {0,0,0,0,0,0,0,0};   // f : cols 128+16w
        v8f acc2 = {0,0,0,0,0,0,0,0};   // g : cols 256+16w
        v8f acc3 = {0,0,0,0,0,0,0,0};   // o : cols 384+16w

        const bf16_t* Abase = (const bf16_t*)(sA + lan15 * KPAD + hi * 16);
        #pragma unroll
        for (int kk = 0; kk < 8; ++kk) {
            // A fragment: lane holds row lan15, 16 contiguous k at k0+16hi
            v8bf a_lo = *(const v8bf*)(Abase + kk * 32);
            v8bf a_hi = *(const v8bf*)(Abase + kk * 32 + 8);
            v16bf afrag = __builtin_shufflevector(a_lo, a_hi,
                0,1,2,3,4,5,6,7,8,9,10,11,12,13,14,15);

            acc0 = __builtin_amdgcn_wmma_f32_16x16x32_bf16(
                       false, afrag, false, Bfr[kk][0], (short)0, acc0, false, false);
            acc1 = __builtin_amdgcn_wmma_f32_16x16x32_bf16(
                       false, afrag, false, Bfr[kk][1], (short)0, acc1, false, false);
            acc2 = __builtin_amdgcn_wmma_f32_16x16x32_bf16(
                       false, afrag, false, Bfr[kk][2], (short)0, acc2, false, false);
            acc3 = __builtin_amdgcn_wmma_f32_16x16x32_bf16(
                       false, afrag, false, Bfr[kk][3], (short)0, acc3, false, false);
        }

        // ============ phase 3: LSTM pointwise update (wave-private columns) =======
        #pragma unroll
        for (int r = 0; r < 8; ++r) {
            const int m = r + hi * 8;           // C/D layout: lanes 16-31 hold rows 8..15
            float gi = acc0[r] + bi;
            float gf = acc1[r] + bff;
            float gg = acc2[r] + bg;
            float go = acc3[r] + bo;

            float si = fast_sigmoid(gi);
            float sf = fast_sigmoid(gf);
            float tg = fast_tanh(gg);
            float so = fast_sigmoid(go);

            float cold = sC[m * H_SZ + colw];
            float cnew = sf * cold + si * tg;
            float hnew = so * fast_tanh(cnew);

            sC[m * H_SZ + colw] = cnew;
            sH[m * H_SZ + colw] = hnew;
        }
        __syncthreads();   // h/c complete -> output copy + next step's phase 1

        // ============ phase 4: coalesced output of h (float4 / b128 stores) =======
        #pragma unroll
        for (int q = 0; q < 2; ++q) {
            int idx = (tid + q * NTHR) * 4;      // element index into [ROWS][H]
            int r = idx >> 7;
            int j = idx & (H_SZ - 1);
            float4 v = *(const float4*)(sH + r * H_SZ + j);
            *(float4*)(out + ((long)(b0 + r) * T_SZ + t) * H_SZ + j) = v;
        }
        // no extra barrier: next phase 1 only READS sH/sC (as does phase 4) and
        // writes sA, which no one reads until after the next barrier.
    }
}

extern "C" void kernel_launch(void* const* d_in, const int* in_sizes, int n_in,
                              void* d_out, int out_size, void* d_ws, size_t ws_size,
                              hipStream_t stream) {
    (void)in_sizes; (void)n_in; (void)out_size; (void)d_ws; (void)ws_size;
    const float* x      = (const float*)d_in[0];
    const float* W_ih   = (const float*)d_in[1];
    const float* W_hh   = (const float*)d_in[2];
    const float* b_ih   = (const float*)d_in[3];
    const float* b_hh   = (const float*)d_in[4];
    const float* W_attn = (const float*)d_in[5];
    const float* b_attn = (const float*)d_in[6];
    float* out = (float*)d_out;

    hipFuncSetAttribute(reinterpret_cast<const void*>(attn_lstm_fused),
                        hipFuncAttributeMaxDynamicSharedMemorySize, SMEM_BYTES);

    attn_lstm_fused<<<B_SZ / ROWS, NTHR, SMEM_BYTES, stream>>>(
        x, W_ih, W_hh, b_ih, b_hh, W_attn, b_attn, out);
}
